// NLBlockND_19662360281679
// MI455X (gfx1250) — matrix operand order repaired and checked by reading, not compiled, not run on hardware
//
#include <hip/hip_runtime.h>
#include <hip/hip_bf16.h>

typedef __bf16 bf16_t;
typedef bf16_t bfx16 __attribute__((ext_vector_type(16)));
typedef bf16_t bfx8  __attribute__((ext_vector_type(8)));
typedef float  v8f   __attribute__((ext_vector_type(8)));

#define WMMA_BF16(A, B, C) \
  __builtin_amdgcn_wmma_f32_16x16x32_bf16(false, (A), false, (B), (short)0, (C), false, false)

#define N_POS   4096      // 4*32*32
#define CIN     80
#define CMID    40
#define CPAD    64        // K padding for bf16 WMMA (2 k-steps of 32)
#define CPAD2   48        // N padding for second GEMM (3 tiles of 16)

__device__ __forceinline__ bfx16 ld16_2(const bf16_t* p0, const bf16_t* p1) {
  bfx8 lo = *(const bfx8*)p0;
  bfx8 hi = *(const bfx8*)p1;
  bfx16 r;
#pragma unroll
  for (int e = 0; e < 8; ++e) { r[e] = lo[e]; r[e + 8] = hi[e]; }
  return r;
}

// ---------------------------------------------------------------------------
// Kernel 1: antialiased linear downsample 64x64 -> 32x32 (taps [1,3,3,1]/8)
// fast/slow: (1,80,4,64,64) f32  ->  xf/xs: [c][t*1024+i*32+j] f32
// ---------------------------------------------------------------------------
__global__ __launch_bounds__(256) void down_kernel(const float* __restrict__ fast,
                                                   const float* __restrict__ slow,
                                                   float* __restrict__ xf,
                                                   float* __restrict__ xs) {
  int idx = blockIdx.x * 256 + threadIdx.x;
  const int per = CIN * 4 * 32 * 32;               // 327680
  if (idx >= 2 * per) return;
  int which = (idx >= per);
  int r = idx - which * per;
  int j = r & 31, i = (r >> 5) & 31, t = (r >> 10) & 3, c = r >> 12;
  const float* src = (which ? slow : fast) + (size_t)(c * 4 + t) * 4096;
  const float wt[4] = {0.125f, 0.375f, 0.375f, 0.125f};
  float acc = 0.f;
#pragma unroll
  for (int a = 0; a < 4; ++a) {
    int y = 2 * i - 1 + a; y = y < 0 ? 0 : (y > 63 ? 63 : y);
    float wa = wt[a];
#pragma unroll
    for (int b = 0; b < 4; ++b) {
      int x = 2 * j - 1 + b; x = x < 0 ? 0 : (x > 63 ? 63 : x);
      acc += wa * wt[b] * src[y * 64 + x];
    }
  }
  (which ? xs : xf)[(size_t)c * N_POS + t * 1024 + i * 32 + j] = acc;
}

// ---------------------------------------------------------------------------
// Kernel 2: projections theta/phi/g (40x80 @ 80x4096), bf16 padded outputs
//   thetaB[n*64+c], phiB[m*64+c] (phi transposed), gT[c*4096+m]
// ---------------------------------------------------------------------------
__global__ __launch_bounds__(256) void proj_kernel(const float* __restrict__ xf,
                                                   const float* __restrict__ g_w,  const float* __restrict__ g_b,
                                                   const float* __restrict__ th_w, const float* __restrict__ th_b,
                                                   const float* __restrict__ ph_w, const float* __restrict__ ph_b,
                                                   bf16_t* __restrict__ thetaB,
                                                   bf16_t* __restrict__ phiB,
                                                   bf16_t* __restrict__ gT) {
  int idx = blockIdx.x * 256 + threadIdx.x;        // 4096*64
  int n = idx >> 6, c = idx & 63;
  if (n >= N_POS) return;
  float th = 0.f, ph = 0.f, gg = 0.f;
  if (c < CMID) {
#pragma unroll 8
    for (int k = 0; k < CIN; ++k) {
      float xv = xf[(size_t)k * N_POS + n];
      th += th_w[c * CIN + k] * xv;
      ph += ph_w[c * CIN + k] * xv;
      gg += g_w[c * CIN + k] * xv;
    }
    th += th_b[c]; ph += ph_b[c]; gg += g_b[c];
  }
  thetaB[(size_t)n * CPAD + c] = (bf16_t)th;
  phiB[(size_t)n * CPAD + c]   = (bf16_t)ph;
  if (c < CPAD2) gT[(size_t)c * N_POS + n] = (bf16_t)gg;
}

// ---------------------------------------------------------------------------
// Kernel 3: flash attention  y = softmax(theta @ phiT) @ g
//  One wave32 per 16-row block (256 waves). GEMM-1 computed transposed:
//  fT tile (A = phiT rows-m, B = theta) so softmax stats are per-lane, and the
//  exp'd probabilities convert in-register into the A-operand of GEMM-2.
// ---------------------------------------------------------------------------
__global__ __launch_bounds__(128) void flash_kernel(const bf16_t* __restrict__ thetaB,
                                                    const bf16_t* __restrict__ phiB,
                                                    const bf16_t* __restrict__ gT,
                                                    float* __restrict__ yOut) {
  const int lane = threadIdx.x & 31;
  const int wv   = threadIdx.x >> 5;
  const int nb   = blockIdx.x * 4 + wv;            // n-block 0..255
  const int nlo  = lane & 15;
  const int h    = lane >> 4;

  // Loop-invariant B operand (theta): B(K=c, N=n) = thetaB[n][c]
  const bf16_t* trow = thetaB + (size_t)(nb * 16 + nlo) * CPAD;
  bfx16 tB0 = ld16_2(trow + h * 8,      trow + h * 8 + 16);
  bfx16 tB1 = ld16_2(trow + 32 + h * 8, trow + 48 + h * 8);

  v8f acc0 = {0,0,0,0,0,0,0,0}, acc1 = {0,0,0,0,0,0,0,0}, acc2 = {0,0,0,0,0,0,0,0};
  float Mrun = -INFINITY, Lrun = 0.f;

  for (int mb = 0; mb < N_POS; mb += 32) {
    // A operand (phiT rows m): lane row m = mb + t*16 + nlo
    const bf16_t* pr0 = phiB + (size_t)(mb + nlo) * CPAD;
    const bf16_t* pr1 = pr0 + 16 * CPAD;
    __builtin_prefetch((const void*)(pr0 + 32 * CPAD), 0, 1);
    bfx16 a00 = ld16_2(pr0 + h * 8,      pr0 + h * 8 + 16);
    bfx16 a01 = ld16_2(pr0 + 32 + h * 8, pr0 + 48 + h * 8);
    bfx16 a10 = ld16_2(pr1 + h * 8,      pr1 + h * 8 + 16);
    bfx16 a11 = ld16_2(pr1 + 32 + h * 8, pr1 + 48 + h * 8);

    v8f f0 = {0,0,0,0,0,0,0,0};
    f0 = WMMA_BF16(a00, tB0, f0);
    f0 = WMMA_BF16(a01, tB1, f0);
    v8f f1 = {0,0,0,0,0,0,0,0};
    f1 = WMMA_BF16(a10, tB0, f1);
    f1 = WMMA_BF16(a11, tB1, f1);

    // Online softmax along m (per lane: column n = nlo; partner lane has other m's)
    float mx = f0[0];
#pragma unroll
    for (int e = 1; e < 8; ++e) mx = fmaxf(mx, f0[e]);
#pragma unroll
    for (int e = 0; e < 8; ++e) mx = fmaxf(mx, f1[e]);
    mx = fmaxf(mx, __shfl_xor(mx, 16, 32));
    float Mnew  = fmaxf(Mrun, mx);
    float scale = __expf(Mrun - Mnew);

    bfx16 pA;                                      // A-operand of GEMM-2, built in-register
    float ls = 0.f;
#pragma unroll
    for (int e = 0; e < 8; ++e) { float p = __expf(f0[e] - Mnew); ls += p; pA[e]     = (bf16_t)p; }
#pragma unroll
    for (int e = 0; e < 8; ++e) { float p = __expf(f1[e] - Mnew); ls += p; pA[e + 8] = (bf16_t)p; }
    ls += __shfl_xor(ls, 16, 32);
    Lrun = Lrun * scale + ls;
    Mrun = Mnew;

    // Rescale accumulators: acc row n_local = r + 8h, scale lives in lane n_local
#pragma unroll
    for (int r = 0; r < 8; ++r) {
      float sr = __shfl(scale, r + 8 * h, 32);
      acc0[r] *= sr; acc1[r] *= sr; acc2[r] *= sr;
    }

    // GEMM-2: y += p @ g ; B(K=m, N=c) from gT[c][m] (contiguous in m)
    const bf16_t* gb = gT + (size_t)nlo * N_POS + mb + h * 8;
    bfx16 gB0 = ld16_2(gb, gb + 16);
    bfx16 gB1 = ld16_2(gb + 16 * N_POS, gb + 16 * N_POS + 16);
    bfx16 gB2 = ld16_2(gb + 32 * N_POS, gb + 32 * N_POS + 16);
    acc0 = WMMA_BF16(pA, gB0, acc0);
    acc1 = WMMA_BF16(pA, gB1, acc1);
    acc2 = WMMA_BF16(pA, gB2, acc2);
  }

  // Normalize and store y (C layout: lane holds y[nb*16 + r + 8h][ct*16 + nlo])
  float inv = 1.0f / Lrun;
#pragma unroll
  for (int r = 0; r < 8; ++r) {
    float ir = __shfl(inv, r + 8 * h, 32);
    size_t row = (size_t)(nb * 16 + r + 8 * h) * CPAD2;
    yOut[row + 0  + nlo] = acc0[r] * ir;
    yOut[row + 16 + nlo] = acc1[r] * ir;
    yOut[row + 32 + nlo] = acc2[r] * ir;
  }
}

// ---------------------------------------------------------------------------
// Kernel 4: wy = wz_w @ y^T + wz_b, plus per-channel sum / sumsq for BN
// ---------------------------------------------------------------------------
__global__ __launch_bounds__(256) void wz_stats_kernel(const float* __restrict__ y,
                                                       const float* __restrict__ wz_w,
                                                       const float* __restrict__ wz_b,
                                                       float* __restrict__ wy,
                                                       float* __restrict__ stats) {
  __shared__ float s1s[256], s2s[256];
  int o = blockIdx.x, tid = threadIdx.x;
  float w[CMID];
#pragma unroll
  for (int c = 0; c < CMID; ++c) w[c] = wz_w[o * CMID + c];
  float wb = wz_b[o], s1 = 0.f, s2 = 0.f;
  for (int n = tid; n < N_POS; n += 256) {
    const float* yr = y + (size_t)n * CPAD2;
    float a = wb;
#pragma unroll
    for (int c = 0; c < CMID; ++c) a += w[c] * yr[c];
    wy[(size_t)o * N_POS + n] = a;
    s1 += a; s2 += a * a;
  }
  s1s[tid] = s1; s2s[tid] = s2;
  __syncthreads();
  for (int st = 128; st > 0; st >>= 1) {
    if (tid < st) { s1s[tid] += s1s[tid + st]; s2s[tid] += s2s[tid + st]; }
    __syncthreads();
  }
  if (tid == 0) { stats[o * 2] = s1s[0]; stats[o * 2 + 1] = s2s[0]; }
}

// ---------------------------------------------------------------------------
// Kernel 5: BN + multiply by slow + linear upsample 32 -> 64
// ---------------------------------------------------------------------------
__global__ __launch_bounds__(256) void bn_mul_up_kernel(const float* __restrict__ wy,
                                                        const float* __restrict__ xs,
                                                        const float* __restrict__ stats,
                                                        const float* __restrict__ bn_g,
                                                        const float* __restrict__ bn_b,
                                                        float* __restrict__ out) {
  int idx = blockIdx.x * 256 + threadIdx.x;
  if (idx >= CIN * 4 * 64 * 64) return;
  int xx = idx & 63, yy = (idx >> 6) & 63, t = (idx >> 12) & 3, o = idx >> 14;
  float mean = stats[o * 2] * (1.f / N_POS);
  float var  = stats[o * 2 + 1] * (1.f / N_POS) - mean * mean;
  float inv  = rsqrtf(var + 1e-5f);
  float gm = bn_g[o], bt = bn_b[o];

  int yH = yy >> 1, xH = xx >> 1;
  int ja, jb, ia, ib; float wya, wyb, wxa, wxb;
  if (yy & 1) { ja = yH; jb = yH + 1 > 31 ? 31 : yH + 1; wya = 0.75f; wyb = 0.25f; }
  else        { ja = yH - 1 < 0 ? 0 : yH - 1; jb = yH;   wya = 0.25f; wyb = 0.75f; }
  if (xx & 1) { ia = xH; ib = xH + 1 > 31 ? 31 : xH + 1; wxa = 0.75f; wxb = 0.25f; }
  else        { ia = xH - 1 < 0 ? 0 : xH - 1; ib = xH;   wxa = 0.25f; wxb = 0.75f; }

  const float* wyp = wy + (size_t)o * N_POS + t * 1024;
  const float* xsp = xs + (size_t)o * N_POS + t * 1024;
#define ZVAL(H, W) ((((wyp[(H)*32+(W)]) - mean) * inv * gm + bt) * xsp[(H)*32+(W)])
  float zaa = ZVAL(ja, ia), zab = ZVAL(ja, ib), zba = ZVAL(jb, ia), zbb = ZVAL(jb, ib);
#undef ZVAL
  out[idx] = wya * (wxa * zaa + wxb * zab) + wyb * (wxa * zba + wxb * zbb);
}

// ---------------------------------------------------------------------------
extern "C" void kernel_launch(void* const* d_in, const int* in_sizes, int n_in,
                              void* d_out, int out_size, void* d_ws, size_t ws_size,
                              hipStream_t stream) {
  const float* fast = (const float*)d_in[0];
  const float* slow = (const float*)d_in[1];
  const float* g_w  = (const float*)d_in[2];
  const float* g_b  = (const float*)d_in[3];
  const float* th_w = (const float*)d_in[4];
  const float* th_b = (const float*)d_in[5];
  const float* ph_w = (const float*)d_in[6];
  const float* ph_b = (const float*)d_in[7];
  const float* wz_w = (const float*)d_in[8];
  const float* wz_b = (const float*)d_in[9];
  const float* bn_g = (const float*)d_in[10];
  const float* bn_b = (const float*)d_in[11];
  float* out = (float*)d_out;

  // workspace layout (all 16B-aligned)
  float* xf    = (float*)d_ws;                 // 80*4096
  float* xs    = xf + CIN * N_POS;             // 80*4096
  float* wy    = xs + CIN * N_POS;             // 80*4096
  float* yOut  = wy + CIN * N_POS;             // 4096*48
  float* stats = yOut + (size_t)N_POS * CPAD2; // 160
  bf16_t* thetaB = (bf16_t*)(stats + 160);
  bf16_t* phiB   = thetaB + (size_t)N_POS * CPAD;
  bf16_t* gT     = phiB   + (size_t)N_POS * CPAD;

  down_kernel<<<(2 * CIN * 4 * 32 * 32 + 255) / 256, 256, 0, stream>>>(fast, slow, xf, xs);
  proj_kernel<<<(N_POS * 64 + 255) / 256, 256, 0, stream>>>(xf, g_w, g_b, th_w, th_b,
                                                            ph_w, ph_b, thetaB, phiB, gT);
  flash_kernel<<<64, 128, 0, stream>>>(thetaB, phiB, gT, yOut);
  wz_stats_kernel<<<CIN, 256, 0, stream>>>(yOut, wz_w, wz_b, wy, stats);
  bn_mul_up_kernel<<<(CIN * 4 * 64 * 64 + 255) / 256, 256, 0, stream>>>(wy, xs, stats,
                                                                        bn_g, bn_b, out);
}
